// RDL_SNNModel_Synaptic_44667659878465
// MI455X (gfx1250) — matrix operand order repaired and checked by reading, not compiled, not run on hardware
//
#include <hip/hip_runtime.h>

// ---------------------------------------------------------------------------
// SNN (5x Linear+Synaptic, T=100, B=512) — persistent fused kernel for gfx1250
//   - f16 WMMA (v_wmma_f32_16x16x32_f16) with f32 accumulation
//   - one workgroup per 16-row batch tile, loops all timesteps internally
//   - spikes exchanged between layers via LDS ping-pong buffers
//   - weights converted once to padded f16 [N][Kpad] in d_ws (L2 resident)
//   - syn/mem state fp32 in d_ws, workgroup-private slices (L2 resident)
// ---------------------------------------------------------------------------

typedef __attribute__((ext_vector_type(16))) _Float16 v16h;
typedef __attribute__((ext_vector_type(8)))  _Float16 v8h;
typedef __attribute__((ext_vector_type(8)))  float    v8f;

union HalfVec { v16h v; v8h h[2]; };

#define T_STEPS   100
#define BATCH     512
#define MTILE     16
#define NWAVES    8
#define NTHREADS  256

struct SnnArgs {
  const float*    data;      // [T, B, 700] fp32
  const _Float16* W[5];      // padded f16 [Npad][Kpad]
  const float*    bias[5];
  const float*    alpha[5];
  const float*    beta[5];
  const float*    thr[5];
  float*          syn[5];    // [B, Npad] fp32
  float*          mem[5];    // [B, Npad] fp32
  float*          out;       // [T, B, 20] fp32
};

// --- weight f32 -> padded f16 conversion ----------------------------------
__global__ void convert_weights(const float* __restrict__ src, _Float16* __restrict__ dst,
                                int Nv, int Ks, int Np, int Kp) {
  int i = blockIdx.x * blockDim.x + threadIdx.x;
  if (i >= Np * Kp) return;
  int n = i / Kp;
  int k = i - n * Kp;
  float v = (n < Nv && k < Ks) ? src[n * Ks + k] : 0.0f;
  dst[i] = (_Float16)v;
}

// --- one fused (Linear + Synaptic) layer for this workgroup's batch tile ---
// KL: padded K, NL: padded N, NVALID: real fan_out, INSTRIDE: LDS row stride
template<int KL, int NL, int NVALID, int INSTRIDE, bool LAST>
__device__ __forceinline__ void layer_fwd(
    const _Float16* __restrict__ Wl,
    const float* __restrict__ biasp,
    const float* __restrict__ alphap,
    const float* __restrict__ betap,
    const float* __restrict__ thrp,
    float* __restrict__ synS,            // state slice base (this batch tile)
    float* __restrict__ memS,
    const _Float16* __restrict__ inb,    // LDS input  (f16 activations/spikes)
    _Float16* __restrict__ outb,         // LDS output (f16 spikes), hidden only
    float* __restrict__ outg,            // d_out, last layer only
    int t, int batch0, int wave, int lane)
{
  constexpr int TILES = NL / 16;                       // 16-col tiles
  constexpr int NT    = (TILES >= NWAVES) ? (TILES / NWAVES) : 1;
  const int nmine = (TILES >= NWAVES) ? NT : ((wave < TILES) ? 1 : 0);
  if (nmine == 0) return;

  const int row = lane & 15;            // A row / B column / C column index
  const int hi  = (lane & 16) ? 1 : 0;  // upper half-wave

  v8f acc[NT];
  #pragma unroll
  for (int s = 0; s < NT; ++s)
    #pragma unroll
    for (int r = 0; r < 8; ++r) acc[s][r] = 0.0f;

  #pragma unroll 2
  for (int kb = 0; kb < KL; kb += 32) {
    // A fragment (16x32 f16): halves 0-7 = K kb+hi*8.., halves 8-15 = +16
    const _Float16* ap = inb + row * INSTRIDE + kb + hi * 8;
    HalfVec ua;
    ua.h[0] = *(const v8h*)(ap);
    ua.h[1] = *(const v8h*)(ap + 16);
    #pragma unroll
    for (int s = 0; s < NT; ++s) {
      if (s < nmine) {
        const int cb = (wave + s * NWAVES) * 16;
        // B fragment (32x16 f16): lane holds W row (cb+row), halves 0-7 = K kb+hi*8.., halves 8-15 = +16
        const _Float16* bp = Wl + (size_t)(cb + row) * KL + kb + hi * 8;
        HalfVec ub;
        ub.h[0] = *(const v8h*)(bp);
        ub.h[1] = *(const v8h*)(bp + 16);
        acc[s] = __builtin_amdgcn_wmma_f32_16x16x32_f16(
            false, ua.v, false, ub.v, (short)0, acc[s], false, false);
      }
    }
  }

  // --- Synaptic epilogue: bias add, state update, spike generation ---------
  const int rbase = hi * 8;
  #pragma unroll
  for (int s = 0; s < NT; ++s) {
    if (s < nmine) {
      const int col = (wave + s * NWAVES) * 16 + row;
      const bool valid = (NVALID == NL) || (col < NVALID);
      const float bias = valid ? biasp[col] : 0.0f;
      const float av = valid ? fminf(fmaxf(alphap[col], 0.0f), 1.0f) : 0.0f;
      const float bv = valid ? fminf(fmaxf(betap[col],  0.0f), 1.0f) : 0.0f;
      const float th = valid ? thrp[col] : 3.0e38f;
      #pragma unroll
      for (int r = 0; r < 8; ++r) {
        const int brow = rbase + r;                 // batch row within tile
        const size_t idx = (size_t)brow * NL + col;
        float sv = synS[idx];
        float mv = memS[idx];
        const float x   = acc[s][r] + bias;
        const float rst = (mv - th) > 0.0f ? th : 0.0f;  // reset * thr (old mem)
        sv = av * sv + x;
        mv = bv * mv + sv - rst;
        synS[idx] = sv;
        memS[idx] = mv;
        const float spk = (mv - th) > 0.0f ? 1.0f : 0.0f;
        if (LAST) {
          if (valid)
            outg[((size_t)t * BATCH + batch0 + brow) * 20 + col] = spk;
        } else {
          outb[brow * 1024 + col] = (_Float16)spk;
        }
      }
    }
  }
}

// --- persistent kernel: one workgroup per 16-row batch tile ----------------
__global__ __launch_bounds__(NTHREADS) void snn_persistent_kernel(SnnArgs a) {
  __shared__ __align__(16) _Float16 buf0[MTILE * 1024];
  __shared__ __align__(16) _Float16 buf1[MTILE * 1024];

  const int tid    = threadIdx.x;
  const int wave   = tid >> 5;
  const int lane   = tid & 31;
  const int batch0 = blockIdx.x * MTILE;

  const int NPAD[5] = {1024, 1024, 1024, 1024, 32};
  float* synS[5];
  float* memS[5];
  #pragma unroll
  for (int l = 0; l < 5; ++l) {
    synS[l] = a.syn[l] + (size_t)batch0 * NPAD[l];
    memS[l] = a.mem[l] + (size_t)batch0 * NPAD[l];
  }

  // zero this workgroup's private state slices (deterministic per launch)
  #pragma unroll
  for (int l = 0; l < 5; ++l) {
    const int n = MTILE * NPAD[l];
    for (int i = tid; i < n; i += NTHREADS) { synS[l][i] = 0.0f; memS[l][i] = 0.0f; }
  }
  __syncthreads();

  for (int t = 0; t < T_STEPS; ++t) {
    // stage input tile: data[t, batch0..+15, 0..699] f32 -> f16 LDS (pad 704)
    for (int i = tid; i < MTILE * 704; i += NTHREADS) {
      const int r = i / 704;
      const int c = i - r * 704;
      float v = (c < 700) ? a.data[((size_t)t * BATCH + batch0 + r) * 700 + c] : 0.0f;
      buf0[r * 704 + c] = (_Float16)v;
    }
    if (t + 1 < T_STEPS)  // hint: pull next step's input toward the caches
      __builtin_prefetch(&a.data[((size_t)(t + 1) * BATCH + batch0) * 700], 0, 1);
    __syncthreads();

    layer_fwd< 704, 1024, 1024,  704, false>(a.W[0], a.bias[0], a.alpha[0], a.beta[0], a.thr[0],
                                             synS[0], memS[0], buf0, buf1, nullptr, t, batch0, wave, lane);
    __syncthreads();
    layer_fwd<1024, 1024, 1024, 1024, false>(a.W[1], a.bias[1], a.alpha[1], a.beta[1], a.thr[1],
                                             synS[1], memS[1], buf1, buf0, nullptr, t, batch0, wave, lane);
    __syncthreads();
    layer_fwd<1024, 1024, 1024, 1024, false>(a.W[2], a.bias[2], a.alpha[2], a.beta[2], a.thr[2],
                                             synS[2], memS[2], buf0, buf1, nullptr, t, batch0, wave, lane);
    __syncthreads();
    layer_fwd<1024, 1024, 1024, 1024, false>(a.W[3], a.bias[3], a.alpha[3], a.beta[3], a.thr[3],
                                             synS[3], memS[3], buf1, buf0, nullptr, t, batch0, wave, lane);
    __syncthreads();
    layer_fwd<1024,   32,   20, 1024, true >(a.W[4], a.bias[4], a.alpha[4], a.beta[4], a.thr[4],
                                             synS[4], memS[4], buf0, nullptr, a.out, t, batch0, wave, lane);
    __syncthreads();
  }
}

// ---------------------------------------------------------------------------
extern "C" void kernel_launch(void* const* d_in, const int* in_sizes, int n_in,
                              void* d_out, int out_size, void* d_ws, size_t ws_size,
                              hipStream_t stream) {
  (void)in_sizes; (void)n_in; (void)out_size; (void)ws_size;

  // layer dims: valid N, source K, padded N, padded K
  static const int Nv[5] = {1024, 1024, 1024, 1024,   20};
  static const int Ks[5] = { 700, 1024, 1024, 1024, 1024};
  static const int Np[5] = {1024, 1024, 1024, 1024,   32};
  static const int Kp[5] = { 704, 1024, 1024, 1024, 1024};

  // workspace layout: [ f16 weights (padded) | f32 syn[5] | f32 mem[5] ]
  _Float16* wsh = (_Float16*)d_ws;
  _Float16* W16[5];
  size_t oh = 0;
  for (int l = 0; l < 5; ++l) { W16[l] = wsh + oh; oh += (size_t)Np[l] * Kp[l]; }
  float* fbase = (float*)((char*)d_ws + oh * sizeof(_Float16));
  float* syn[5];
  float* mem[5];
  size_t fo = 0;
  for (int l = 0; l < 5; ++l) { syn[l] = fbase + fo; fo += (size_t)BATCH * Np[l]; }
  for (int l = 0; l < 5; ++l) { mem[l] = fbase + fo; fo += (size_t)BATCH * Np[l]; }

  // input order: data, then per layer {W, b, alpha, beta, thr}, then timestep
  for (int l = 0; l < 5; ++l) {
    const float* src = (const float*)d_in[1 + 5 * l];
    const int tot = Np[l] * Kp[l];
    convert_weights<<<(tot + 255) / 256, 256, 0, stream>>>(src, W16[l], Nv[l], Ks[l], Np[l], Kp[l]);
  }

  SnnArgs a;
  a.data = (const float*)d_in[0];
  for (int l = 0; l < 5; ++l) {
    a.W[l]     = W16[l];
    a.bias[l]  = (const float*)d_in[2 + 5 * l];
    a.alpha[l] = (const float*)d_in[3 + 5 * l];
    a.beta[l]  = (const float*)d_in[4 + 5 * l];
    a.thr[l]   = (const float*)d_in[5 + 5 * l];
    a.syn[l]   = syn[l];
    a.mem[l]   = mem[l];
  }
  a.out = (float*)d_out;

  snn_persistent_kernel<<<BATCH / MTILE, NTHREADS, 0, stream>>>(a);
}
